// HungarianMatcher_78786880078115
// MI455X (gfx1250) — matrix operand order repaired and checked by reading, not compiled, not run on hardware
//
#include <hip/hip_runtime.h>
#include <hip/hip_bf16.h>

typedef __attribute__((ext_vector_type(16))) _Float16 v16h;
typedef __attribute__((ext_vector_type(8)))  _Float16 v8h;
typedef __attribute__((ext_vector_type(8)))  float    v8f;

namespace {
constexpr int kB   = 8;
constexpr int kQ   = 100;
constexpr int kT   = 50;
constexpr int kHW  = 256 * 256;          // 65536
constexpr int kMT  = 7;                  // ceil(Q/16)
constexpr int kMP  = kMT * 16;           // 112 padded Q
constexpr int kNP  = 64;                 // padded T (4 n-tiles of 16)
constexpr int kKS  = 32;                 // K slices (grid.x)
constexpr int kKLEN = kHW / kKS;         // 2048 K per slice
constexpr float kLog2E = 1.44269504088896340f;   // log2(e)
constexpr float kLn2   = 0.69314718055994531f;   // ln(2)
// workspace layout (in floats, then an f16 region)
constexpr size_t kG1Off   = 0;
constexpr size_t kG2Off   = kG1Off + (size_t)kB * kMP * kNP;
constexpr size_t kRnegOff = kG2Off + (size_t)kB * kMP * kNP;
constexpr size_t kRsOff   = kRnegOff + (size_t)kB * kMP;
constexpr size_t kRtOff   = kRsOff + (size_t)kB * kMP;
constexpr size_t kWsFloats = kRtOff + (size_t)kB * kNP;   // 116992 (16B aligned)
constexpr size_t kTF16Elems = (size_t)kB * kT * kHW;      // 26.2M f16
}

// ---------------------------------------------------------------------------
// Pass 1: stream-convert binary t (int32) -> exact f16 in workspace, and
// reduce per-row sums (Sigma t) with a wave32 shuffle tree + one f32 atomic.
// grid: (kHW / (256*8) = 32, kT, kB), 256 threads, 8 elements/thread.
// ---------------------------------------------------------------------------
__global__ __launch_bounds__(256)
void matcher_tcast(const int* __restrict__ tg, _Float16* __restrict__ tf,
                   float* __restrict__ rt)
{
  const int b = blockIdx.z;
  const int n = blockIdx.y;
  const int base = (blockIdx.x * 256 + threadIdx.x) * 8;
  const size_t off = (size_t)(b * kT + n) * kHW + base;

  const int4* p = reinterpret_cast<const int4*>(tg + off);
  const int4 a = p[0], c = p[1];
  const int v[8] = {a.x, a.y, a.z, a.w, c.x, c.y, c.z, c.w};

  v8h h;
  float s = 0.0f;
#pragma unroll
  for (int j = 0; j < 8; ++j) {
    const float f = (float)v[j];
    s += f;
    h[j] = (_Float16)f;
  }
  *reinterpret_cast<v8h*>(tf + off) = h;   // 16B store -> global_store_b128

  // wave32 reduction of the row-sum contribution
#pragma unroll
  for (int d = 16; d > 0; d >>= 1) s += __shfl_xor(s, d, 32);
  if ((threadIdx.x & 31) == 0) atomicAdd(&rt[b * kNP + n], s);
}

// ---------------------------------------------------------------------------
// Pass 2: per (k-slice, m-tile, batch) workgroup of 4 wave32s. Each wave owns
// an interleaved K stream and accumulates x.t^T and sig(x).t^T for all four
// 16-wide N tiles with v_wmma_f32_16x16x32_f16. The transcendental chain
// uses the raw gfx1250 v_exp_f32 / v_log_f32 (safe here: exp argument <= 0,
// log argument in [1,2]) to avoid libm denorm/overflow fixup sequences.
// ---------------------------------------------------------------------------
__global__ __launch_bounds__(128)
void matcher_gemm(const float* __restrict__ x, const _Float16* __restrict__ tf,
                  float* __restrict__ G1, float* __restrict__ G2,
                  float* __restrict__ rneg, float* __restrict__ rs)
{
  const int slice = blockIdx.x;
  const int mt    = blockIdx.y;
  const int b     = blockIdx.z;
  const int tid   = threadIdx.x;
  const int wave  = tid >> 5;
  const int lane  = tid & 31;
  const int half  = lane >> 4;   // which 16-lane group
  const int l     = lane & 15;

  // A-matrix row this lane feeds (clamped tail rows are never read back).
  const int q  = mt * 16 + l;
  const int qc = q < kQ ? q : (kQ - 1);
  const float* xrow = x + ((size_t)(b * kQ + qc) * kHW);

  // B-matrix rows (t columns, pre-converted f16), one per n-tile.
  const _Float16* trow[4];
#pragma unroll
  for (int nt = 0; nt < 4; ++nt) {
    const int n  = nt * 16 + l;
    const int nc = n < kT ? n : (kT - 1);
    trow[nt] = tf + ((size_t)(b * kT + nc) * kHW);
  }

  v8f acc1[4] = {};                 // x . t^T      per n-tile
  v8f acc2[4] = {};                 // sig(x) . t^T per n-tile
  float sum_sp = 0.0f, sum_s = 0.0f;

  const int kbeg = slice * kKLEN + wave * 32;
  const int kend = slice * kKLEN + kKLEN;

  for (int kb = kbeg; kb < kend; kb += 128) {
    // ---- 16 x values per the 16-bit A-matrix VGPR striping:
    // half 0: K = kb+0..7 and kb+16..23 ; half 1: K = kb+8..15 and kb+24..31
    __builtin_prefetch(xrow + kb + 512, 0, 1);   // global_prefetch_b8
    float xv[16];
    {
      const float4* p0 = reinterpret_cast<const float4*>(xrow + kb + half * 8);
      const float4* p1 = reinterpret_cast<const float4*>(xrow + kb + 16 + half * 8);
      const float4 a0 = p0[0], a1 = p0[1];
      const float4 b0 = p1[0], b1 = p1[1];
      xv[0]  = a0.x; xv[1]  = a0.y; xv[2]  = a0.z; xv[3]  = a0.w;
      xv[4]  = a1.x; xv[5]  = a1.y; xv[6]  = a1.z; xv[7]  = a1.w;
      xv[8]  = b0.x; xv[9]  = b0.y; xv[10] = b0.z; xv[11] = b0.w;
      xv[12] = b1.x; xv[13] = b1.y; xv[14] = b1.z; xv[15] = b1.w;
    }

    // ---- sigmoid / softplus via raw v_exp_f32 / v_log_f32
    v16h ax, as;
#pragma unroll
    for (int j = 0; j < 16; ++j) {
      const float xf  = xv[j];
      // e = exp(-|x|) = exp2(-|x| * log2e); arg <= 0 so result in (0,1]
      const float e   = __builtin_amdgcn_exp2f(-__builtin_fabsf(xf) * kLog2E);
      const float inv = __builtin_amdgcn_rcpf(1.0f + e);
      const float sg  = (xf >= 0.0f) ? inv : e * inv;            // sigmoid(x)
      // softplus(x) = max(x,0) + ln(1+e); 1+e in [1,2] -> raw v_log_f32 safe
      const float sp  = __builtin_fmaxf(xf, 0.0f) +
                        __builtin_amdgcn_logf(1.0f + e) * kLn2;
      sum_s  += sg;
      sum_sp += sp;
      ax[j] = (_Float16)xf;
      as[j] = (_Float16)sg;
    }

    // ---- per n-tile: B fragment matches A striping
    // (half 0 -> K = kb+0..7,16..23, half 1 -> K = kb+8..15,24..31): two b128 loads
#pragma unroll
    for (int nt = 0; nt < 4; ++nt) {
      const v8h blo = *reinterpret_cast<const v8h*>(trow[nt] + kb + half * 8);
      const v8h bhi = *reinterpret_cast<const v8h*>(trow[nt] + kb + 16 + half * 8);
      const v16h bt = __builtin_shufflevector(blo, bhi, 0,1,2,3,4,5,6,7,8,9,10,11,12,13,14,15);
      acc1[nt] = __builtin_amdgcn_wmma_f32_16x16x32_f16(
          false, ax, false, bt, (short)0, acc1[nt], false, false);
      acc2[nt] = __builtin_amdgcn_wmma_f32_16x16x32_f16(
          false, as, false, bt, (short)0, acc2[nt], false, false);
    }
  }

  // ---- fold the two lane halves (complementary K) and flush row sums
  sum_sp += __shfl_xor(sum_sp, 16, 32);
  sum_s  += __shfl_xor(sum_s, 16, 32);
  if (half == 0 && q < kQ) {
    atomicAdd(&rneg[b * kMP + q], sum_sp);
    atomicAdd(&rs[b * kMP + q], sum_s);
  }

  // ---- flush the 16x16 f32 accumulator tiles (C/D layout: VGPR r holds
  // row r + 8*half, column l) into workspace, summing across K slices
#pragma unroll
  for (int nt = 0; nt < 4; ++nt) {
    const int col = nt * 16 + l;
#pragma unroll
    for (int r = 0; r < 8; ++r) {
      const int row = mt * 16 + r + 8 * half;
      const size_t idx = ((size_t)b * kMP + row) * kNP + col;
      atomicAdd(&G1[idx], acc1[nt][r]);
      atomicAdd(&G2[idx], acc2[nt][r]);
    }
  }
}

// ---------------------------------------------------------------------------
// Pass 3: C = (Sigma softplus(x) - x.t)/HW + 1 - (2*sig(x).t + 1)/(Ss + St + 1)
// ---------------------------------------------------------------------------
__global__ __launch_bounds__(256)
void matcher_combine(const float* __restrict__ G1, const float* __restrict__ G2,
                     const float* __restrict__ rneg, const float* __restrict__ rs,
                     const float* __restrict__ rt, float* __restrict__ out)
{
  const int i = blockIdx.x * blockDim.x + threadIdx.x;
  if (i >= kB * kQ * kT) return;
  const int t = i % kT;
  const int q = (i / kT) % kQ;
  const int b = i / (kQ * kT);
  const size_t idx = ((size_t)b * kMP + q) * kNP + t;
  const float g1 = G1[idx];
  const float g2 = G2[idx];
  const float ce   = (rneg[b * kMP + q] - g1) * (1.0f / (float)kHW);
  const float dice = 1.0f - (2.0f * g2 + 1.0f) /
                            (rs[b * kMP + q] + rt[b * kNP + t] + 1.0f);
  out[i] = ce + dice;
}

extern "C" void kernel_launch(void* const* d_in, const int* in_sizes, int n_in,
                              void* d_out, int out_size, void* d_ws, size_t ws_size,
                              hipStream_t stream) {
  const float* x  = (const float*)d_in[0];   // pred_masks [8,100,256,256] f32
  const int*   tg = (const int*)d_in[1];     // tgt_masks  [8,50,256,256] i32

  float* ws   = (float*)d_ws;
  float* G1   = ws + kG1Off;
  float* G2   = ws + kG2Off;
  float* rneg = ws + kRnegOff;
  float* rs   = ws + kRsOff;
  float* rt   = ws + kRtOff;
  _Float16* tf16 = reinterpret_cast<_Float16*>(ws + kWsFloats);

  // zero only the accumulator region (f16 region is fully overwritten)
  hipMemsetAsync(d_ws, 0, kWsFloats * sizeof(float), stream);

  dim3 gridT(kHW / (256 * 8), kT, kB);       // (32, 50, 8)
  matcher_tcast<<<gridT, 256, 0, stream>>>(tg, tf16, rt);

  dim3 gridG(kKS, kMT, kB);                  // (32, 7, 8)
  matcher_gemm<<<gridG, 128, 0, stream>>>(x, tf16, G1, G2, rneg, rs);

  const int total = kB * kQ * kT;
  matcher_combine<<<(total + 255) / 256, 256, 0, stream>>>(
      G1, G2, rneg, rs, rt, (float*)d_out);
}
